// GraphClsGGNN_52621939310628
// MI455X (gfx1250) — matrix operand order repaired and checked by reading, not compiled, not run on hardware
//
#include <hip/hip_runtime.h>

#define DD 128
#define GG 64
#define NCLSS 2
#define GSTEPS 5
#define KBLK 4   // K=128 -> 4 blocks of 32

typedef __attribute__((ext_vector_type(16))) _Float16 v16h;
typedef __attribute__((ext_vector_type(8)))  _Float16 v8h;
typedef __attribute__((ext_vector_type(8)))  float    v8f;

// ---------------------------------------------------------------------------
// WMMA GEMM: C[M x Ncols] = A[M x 128 (f16)] * B(preswizzled f16) + bias
// One wave -> one 16x16 C tile; 8 waves/block -> 128 cols; grid.y batches
// column groups of 128. M multiple of 16 (N=50000=3125*16) -> EXEC all-ones.
//
// The 16x128 A tile is staged into LDS once per block with
// global_load_async_to_lds_b128 (ASYNCcnt) and shared by all 8 waves; each
// wave then builds its A fragments from LDS (ds loads).
//
// B pre-swizzle: idx = ((ct*KBLK + kb)*512) + lane*16 + i
//   lane<16 : b[i] = B[kb*32 + (i<8 ? i : 8+i)][ct*16 + lane]
//   lane>=16: b[i] = B[kb*32 + (i<8 ? 8+i : 16+i)][ct*16 + lane-16]
// A fragment per ISA 16-bit A 16x32 table:
//   lanes0-15: K 0..7 / 16..23 ; lanes16-31: K 8..15 / 24..31
// ---------------------------------------------------------------------------
__global__ __launch_bounds__(256)
void ggnn_wmma_gemm(const _Float16* __restrict__ A,
                    const _Float16* __restrict__ Bsw,
                    const float* __restrict__ bias,
                    float* __restrict__ C, int ldc) {
  __shared__ __align__(16) _Float16 atile[16 * DD];   // 4 KB

  const int t    = threadIdx.x;
  const int lane = t & 31;
  const int wave = t >> 5;
  const int m0   = blockIdx.x * 16;
  const int ct   = blockIdx.y * 8 + wave;      // global 16-col tile index
  const int hi   = lane >> 4;                  // lane half select
  const int col  = ct * 16 + (lane & 15);

  // ---- async stage A tile: thread t copies 16B; linear layout matches
  //      atile[row*128 + col] since t*8 == (t>>4)*128 + (t&15)*8 ----
  {
    const _Float16* gsrc = A + (size_t)m0 * DD + t * 8;
    unsigned ldsoff = (unsigned)(unsigned long long)(const void*)&atile[t * 8];
    asm volatile("global_load_async_to_lds_b128 %0, %1, off"
                 :: "v"(ldsoff), "v"(gsrc)
                 : "memory");
    asm volatile("s_wait_asynccnt 0x0" ::: "memory");
  }
  __syncthreads();

  const _Float16* arow = &atile[(lane & 15) * DD];
  const _Float16* bp   = Bsw + (size_t)ct * KBLK * 512 + lane * 16;

  v8f acc = {};
#pragma unroll
  for (int kb = 0; kb < KBLK; ++kb) {
    const int kbase = kb * 32;
    v8h alo = *(const v8h*)(arow + kbase + (hi ? 8 : 0));
    v8h aup = *(const v8h*)(arow + kbase + (hi ? 24 : 16));
    v16h afrag;
#pragma unroll
    for (int i = 0; i < 8; ++i) { afrag[i] = alo[i]; afrag[8 + i] = aup[i]; }
    v16h bfrag = *(const v16h*)(bp + (size_t)kb * 512);
    acc = __builtin_amdgcn_wmma_f32_16x16x32_f16(
        /*neg_a=*/false, afrag, /*neg_b=*/false, bfrag,
        /*c_mod=*/(short)0, acc, /*reuse_a=*/false, /*reuse_b=*/false);
  }

  const float bv = bias ? bias[col] : 0.0f;
#pragma unroll
  for (int r = 0; r < 8; ++r) {
    const int orow = m0 + (hi ? 8 : 0) + r;   // C layout: VGPR r -> M=r / M=8+r
    C[(size_t)orow * ldc + col] = acc[r] + bv;
  }
}

// Pre-swizzle an f32 weight matrix into WMMA B-fragment order (f16).
// trans==0: logical B[k][col] = src[k*Ncols + col]
// trans==1: logical B[k][col] = src[col*Kdim + k]   (for wih^T / whh^T)
__global__ void ggnn_swizzle_b(const float* __restrict__ src, _Float16* __restrict__ dst,
                               int Kdim, int Ncols, int trans) {
  const int idx = blockIdx.x * blockDim.x + threadIdx.x;
  const int kblocks = Kdim >> 5;
  const int total = (Ncols >> 4) * kblocks * 512;
  if (idx >= total) return;
  const int i    = idx & 15;
  const int lane = (idx >> 4) & 31;
  const int kb   = (idx >> 9) % kblocks;
  const int ct   = idx / (512 * kblocks);
  const int col  = ct * 16 + (lane & 15);
  const int k    = kb * 32 + ((lane >> 4) ? 8 : 0) + (i & 7) + ((i >> 3) << 4);
  const float v  = trans ? src[(size_t)col * Kdim + k] : src[(size_t)k * Ncols + col];
  dst[idx] = (_Float16)v;
}

// h0 = [feat | zeros], plus f16 copy
__global__ void ggnn_init_h(const float* __restrict__ feat, float* __restrict__ h,
                            _Float16* __restrict__ h16, int N) {
  const int idx = blockIdx.x * blockDim.x + threadIdx.x;
  if (idx >= N * DD) return;
  const int n = idx / DD, d = idx - n * DD;
  const float v = (d < 32) ? feat[n * 32 + d] : 0.0f;
  h[idx] = v; h16[idx] = (_Float16)v;
}

__global__ void ggnn_zero_f32(float* __restrict__ p, int n) {
  const int i = blockIdx.x * blockDim.x + threadIdx.x;
  if (i < n) p[i] = 0.0f;
}

__global__ void ggnn_f32_to_f16(const float* __restrict__ s, _Float16* __restrict__ d, int n) {
  const int i = blockIdx.x * blockDim.x + threadIdx.x;
  if (i < n) d[i] = (_Float16)s[i];
}

// per-edge gather of hW[src, etype, :] (128 f32) and atomic scatter into a[dst, :]
__global__ void ggnn_edge_scatter(const float* __restrict__ hW,
                                  const int* __restrict__ src, const int* __restrict__ dst,
                                  const int* __restrict__ et, float* __restrict__ a, int E) {
  const int t = blockIdx.x * blockDim.x + threadIdx.x;
  const int e = t >> 5, lane = t & 31;
  if (e >= E) return;
  const int s = src[e], d = dst[e], k = et[e];
  const float4 v = ((const float4*)(hW + (size_t)s * (4 * DD) + (size_t)k * DD))[lane];
  float* ap = a + (size_t)d * DD + lane * 4;
  __hip_atomic_fetch_add(ap + 0, v.x, __ATOMIC_RELAXED, __HIP_MEMORY_SCOPE_AGENT);
  __hip_atomic_fetch_add(ap + 1, v.y, __ATOMIC_RELAXED, __HIP_MEMORY_SCOPE_AGENT);
  __hip_atomic_fetch_add(ap + 2, v.z, __ATOMIC_RELAXED, __HIP_MEMORY_SCOPE_AGENT);
  __hip_atomic_fetch_add(ap + 3, v.w, __ATOMIC_RELAXED, __HIP_MEMORY_SCOPE_AGENT);
}

// GRUCell elementwise update (in-place on h), optional trailing ReLU
__global__ void ggnn_gru_update(const float* __restrict__ gi, const float* __restrict__ gh,
                                float* __restrict__ h, _Float16* __restrict__ h16,
                                int N, int relu) {
  const int idx = blockIdx.x * blockDim.x + threadIdx.x;
  if (idx >= N * DD) return;
  const int n = idx / DD, d = idx - n * DD;
  const size_t b = (size_t)n * (3 * DD);
  const float ir = gi[b + d], iz = gi[b + DD + d], in = gi[b + 2 * DD + d];
  const float hr = gh[b + d], hz = gh[b + DD + d], hn = gh[b + 2 * DD + d];
  const float r = 1.0f / (1.0f + __expf(-(ir + hr)));
  const float z = 1.0f / (1.0f + __expf(-(iz + hz)));
  const float nn = tanhf(in + r * hn);
  float hv = (1.0f - z) * nn + z * h[idx];
  if (relu) hv = fmaxf(hv, 0.0f);
  h[idx] = hv; h16[idx] = (_Float16)hv;
}

// ---- attention pooling ----
__device__ __forceinline__ unsigned ggnn_f2mono(float x) {
  unsigned b = __float_as_uint(x);
  return (b & 0x80000000u) ? ~b : (b | 0x80000000u);
}
__device__ __forceinline__ float ggnn_mono2f(unsigned m) {
  unsigned b = (m & 0x80000000u) ? (m & 0x7FFFFFFFu) : ~m;
  return __uint_as_float(b);
}

__global__ void ggnn_pool_init(unsigned* __restrict__ gmax, float* __restrict__ denom,
                               float* __restrict__ hg) {
  const int i = blockIdx.x * blockDim.x + threadIdx.x;
  if (i < GG) { gmax[i] = 0x00800000u /* mono(-FLT_MAX) */; denom[i] = 0.0f; }
  if (i < GG * DD) hg[i] = 0.0f;
}

// one wave32 per node: gate score + segment max
__global__ void ggnn_gate_score(const float* __restrict__ h, const float* __restrict__ gw,
                                const float* __restrict__ gb, const int* __restrict__ gid,
                                float* __restrict__ g, unsigned* __restrict__ gmax, int N) {
  const int node = blockIdx.x * 8 + (threadIdx.x >> 5);
  const int lane = threadIdx.x & 31;
  if (node >= N) return;
  float s = 0.0f;
  for (int d = lane; d < DD; d += 32) s += h[(size_t)node * DD + d] * gw[d];
  for (int off = 16; off > 0; off >>= 1) s += __shfl_down(s, off, 32);
  if (lane == 0) {
    const float sc = s + gb[0];
    g[node] = sc;
    atomicMax(&gmax[gid[node]], ggnn_f2mono(sc));
  }
}

__global__ void ggnn_pool_exp(const float* __restrict__ g, const unsigned* __restrict__ gmax,
                              const int* __restrict__ gid, float* __restrict__ e,
                              float* __restrict__ denom, int N) {
  const int n = blockIdx.x * blockDim.x + threadIdx.x;
  if (n >= N) return;
  const int gi = gid[n];
  const float x = __expf(g[n] - ggnn_mono2f(gmax[gi]));
  e[n] = x;
  __hip_atomic_fetch_add(&denom[gi], x, __ATOMIC_RELAXED, __HIP_MEMORY_SCOPE_AGENT);
}

__global__ void ggnn_pool_wsum(const float* __restrict__ h, const float* __restrict__ e,
                               const float* __restrict__ denom, const int* __restrict__ gid,
                               float* __restrict__ hg, int N) {
  const int n = blockIdx.x, d = threadIdx.x;
  const int gi = gid[n];
  const float alpha = e[n] / denom[gi];
  __hip_atomic_fetch_add(&hg[(size_t)gi * DD + d], alpha * h[(size_t)n * DD + d],
                         __ATOMIC_RELAXED, __HIP_MEMORY_SCOPE_AGENT);
}

__global__ void ggnn_classify(const float* __restrict__ hg, const float* __restrict__ fw,
                              const float* __restrict__ fb, float* __restrict__ out) {
  const int t = threadIdx.x;
  if (t >= GG * NCLSS) return;
  const int g = t >> 1, c = t & 1;
  float s = fb[c];
  for (int d = 0; d < DD; ++d) s += hg[g * DD + d] * fw[d * NCLSS + c];
  out[t] = s;
}

// ---------------------------------------------------------------------------
extern "C" void kernel_launch(void* const* d_in, const int* in_sizes, int n_in,
                              void* d_out, int out_size, void* d_ws, size_t ws_size,
                              hipStream_t stream) {
  const float* feat  = (const float*)d_in[0];
  const int*   src   = (const int*)d_in[1];
  const int*   dstv  = (const int*)d_in[2];
  const int*   etype = (const int*)d_in[3];
  const int*   gid   = (const int*)d_in[4];
  const float* W[2]   = {(const float*)d_in[5],  (const float*)d_in[11]};
  const float* bW[2]  = {(const float*)d_in[6],  (const float*)d_in[12]};
  const float* wih[2] = {(const float*)d_in[7],  (const float*)d_in[13]};
  const float* whh[2] = {(const float*)d_in[8],  (const float*)d_in[14]};
  const float* bih[2] = {(const float*)d_in[9],  (const float*)d_in[15]};
  const float* bhh[2] = {(const float*)d_in[10], (const float*)d_in[16]};
  const float* gate_w = (const float*)d_in[17];
  const float* gate_b = (const float*)d_in[18];
  const float* fc_w   = (const float*)d_in[19];
  const float* fc_b   = (const float*)d_in[20];

  const int E = in_sizes[1];
  const int N = in_sizes[4];          // 50000, a multiple of 16

  // workspace carve
  char* p = (char*)d_ws;
  auto carve = [&](size_t bytes) -> void* {
    void* r = (void*)p; p += (bytes + 255) & ~(size_t)255; return r;
  };
  float*    h    = (float*)   carve((size_t)N * DD * 4);
  _Float16* h16  = (_Float16*)carve((size_t)N * DD * 2);
  float*    aagg = (float*)   carve((size_t)N * DD * 4);
  _Float16* a16  = (_Float16*)carve((size_t)N * DD * 2);
  float*    hW   = (float*)   carve((size_t)N * 4 * DD * 4);   // [N][K*D]
  float*    gib  = (float*)   carve((size_t)N * 3 * DD * 4);
  float*    ghb  = (float*)   carve((size_t)N * 3 * DD * 4);
  _Float16* Wsw   = (_Float16*)carve((size_t)4 * 8 * KBLK * 512 * 2);  // 4 etypes
  _Float16* wihsw = (_Float16*)carve((size_t)24 * KBLK * 512 * 2);
  _Float16* whhsw = (_Float16*)carve((size_t)24 * KBLK * 512 * 2);
  float*    gsc   = (float*)   carve((size_t)N * 4);
  float*    ebuf  = (float*)   carve((size_t)N * 4);
  unsigned* gmax  = (unsigned*)carve(GG * 4);
  float*    denom = (float*)   carve(GG * 4);
  float*    hg    = (float*)   carve(GG * DD * 4);

  const int ND  = N * DD;
  const int nd_blocks = (ND + 255) / 256;
  const int mtiles = N / 16;

  ggnn_init_h<<<nd_blocks, 256, 0, stream>>>(feat, h, h16, N);

  for (int L = 0; L < 2; ++L) {
    // weight prep: f32 -> f16, WMMA B-fragment order
    for (int k = 0; k < 4; ++k)
      ggnn_swizzle_b<<<(8 * KBLK * 512 + 255) / 256, 256, 0, stream>>>(
          W[L] + (size_t)k * DD * DD, Wsw + (size_t)k * 8 * KBLK * 512, DD, DD, 0);
    ggnn_swizzle_b<<<(24 * KBLK * 512 + 255) / 256, 256, 0, stream>>>(wih[L], wihsw, DD, 3 * DD, 1);
    ggnn_swizzle_b<<<(24 * KBLK * 512 + 255) / 256, 256, 0, stream>>>(whh[L], whhsw, DD, 3 * DD, 1);

    for (int s = 0; s < GSTEPS; ++s) {
      // hW[n, k*D+f] = h @ Wbig + b   (one GEMM, Ncols=512)
      ggnn_wmma_gemm<<<dim3(mtiles, 4), 256, 0, stream>>>(h16, Wsw, bW[L], hW, 4 * DD);
      // a = segment_sum over edges
      ggnn_zero_f32<<<nd_blocks, 256, 0, stream>>>(aagg, ND);
      ggnn_edge_scatter<<<((size_t)E * 32 + 255) / 256, 256, 0, stream>>>(hW, src, dstv, etype, aagg, E);
      ggnn_f32_to_f16<<<nd_blocks, 256, 0, stream>>>(aagg, a16, ND);
      // GRU gates
      ggnn_wmma_gemm<<<dim3(mtiles, 3), 256, 0, stream>>>(a16, wihsw, bih[L], gib, 3 * DD);
      ggnn_wmma_gemm<<<dim3(mtiles, 3), 256, 0, stream>>>(h16, whhsw, bhh[L], ghb, 3 * DD);
      // GRU elementwise (ReLU folded into last step of each layer)
      ggnn_gru_update<<<nd_blocks, 256, 0, stream>>>(gib, ghb, h, h16, N, (s == GSTEPS - 1) ? 1 : 0);
    }
  }

  // global attention pooling + classifier
  ggnn_pool_init<<<(GG * DD + 255) / 256, 256, 0, stream>>>(gmax, denom, hg);
  ggnn_gate_score<<<(N + 7) / 8, 256, 0, stream>>>(h, gate_w, gate_b, gid, gsc, gmax, N);
  ggnn_pool_exp<<<(N + 255) / 256, 256, 0, stream>>>(gsc, gmax, gid, ebuf, denom, N);
  ggnn_pool_wsum<<<N, DD, 0, stream>>>(h, ebuf, denom, gid, hg, N);
  ggnn_classify<<<1, 128, 0, stream>>>(hg, fc_w, fc_b, (float*)d_out);
}
